// HAR_Mamba_59519656788059
// MI455X (gfx1250) — compile-verified, matrix operand order checked by reading, not run on hardware
//
#include <hip/hip_runtime.h>
#include <hip/hip_bf16.h>

// ---------------- types (gfx1250, wave32) ----------------
typedef __bf16 bf16;
typedef __attribute__((ext_vector_type(16))) __bf16 v16bf;
typedef __attribute__((ext_vector_type(8)))  __bf16 v8bf;
typedef __attribute__((ext_vector_type(8)))  float  v8f;
typedef __attribute__((ext_vector_type(4)))  unsigned int v4u;
typedef __attribute__((ext_vector_type(8)))  int v8i;
typedef __attribute__((ext_vector_type(4)))  int v4i;

#define B_   128
#define L_   1024
#define IND  64
#define H_   256
#define DI   512        // D_INNER
#define NS   16         // state dim N
#define RD   16         // R
#define BL   (B_ * L_)  // 131072 rows

// LDS B-tile: rows are 32 bf16 (64B) + 16B TDM pad -> 80B pitch, 16B aligned,
// and lid*20 dwords mod 64 banks is conflict-free for the 16-lane groups.
#define BROW_PITCH 80

union FragBF { v16bf v; v8bf h[2]; };

__device__ __forceinline__ v8bf cvt8_f32_bf16(const float* __restrict__ p) {
  v8bf r;
#pragma unroll
  for (int i = 0; i < 8; ++i) r[i] = (__bf16)p[i];
  return r;
}

// -----------------------------------------------------------------
// TDM: async 2D bf16 tile load Global -> LDS, with LDS row padding.
//   tile: tile_n rows x 32 elems; global row stride = tensorK elems.
//   pad: after every 16 dwords (one 64B row) insert 4 dwords -> 80B pitch.
// D# layout per CDNA5 ISA 8.3/8.4 (group0: lds/global addr, type=2;
// group1: data_size=2B, pad_enable, dims/strides).
// This toolchain's builtin takes 6 args:
//   (uint32x4 g0, int32x8 g1, int32x4 g2, int32x4 g3, int32x8 extra, i32 cpol)
// -----------------------------------------------------------------
__device__ __forceinline__ void tdm_load_b_tile(unsigned lds_off,
                                                const bf16* gptr,
                                                unsigned tile_n,
                                                unsigned tensorK,
                                                unsigned tensorN) {
#if __has_builtin(__builtin_amdgcn_tensor_load_to_lds)
  unsigned long long ga = (unsigned long long)(size_t)gptr;
  v4u g0;
  g0[0] = 1u;                                   // count=1 (user descriptor)
  g0[1] = lds_off;                              // LDS byte address
  g0[2] = (unsigned)(ga & 0xffffffffu);         // global_addr[31:0]
  g0[3] = (unsigned)((ga >> 32) & 0x01ffffffu)  // global_addr[56:32]
          | (2u << 30);                         // type=2 ("image")
  v8i g1;
  // data_size=1(2B) | pad_enable | pad_interval=3(16dw) | pad_amount=3(4dw)
  g1[0] = (int)0x06D10000;
  g1[1] = (int)((tensorK & 0xffffu) << 16);                       // tensor_dim0 lo
  g1[2] = (int)(((tensorK >> 16) & 0xffffu) |
                ((tensorN & 0xffffu) << 16));                     // td0 hi | td1 lo
  g1[3] = (int)(((tensorN >> 16) & 0xffffu) | (32u << 16));       // td1 hi | tile_dim0=32
  g1[4] = (int)(tile_n & 0xffffu);                                // tile_dim1 (tile_dim2=0)
  g1[5] = (int)tensorK;                                           // tensor_dim0_stride[31:0]
  g1[6] = 0;                                                      // stride0 hi | stride1 lo
  g1[7] = 0;
  v4i z4; z4[0] = 0; z4[1] = 0; z4[2] = 0; z4[3] = 0;
  v8i z8; z8[0] = 0; z8[1] = 0; z8[2] = 0; z8[3] = 0;
  z8[4] = 0; z8[5] = 0; z8[6] = 0; z8[7] = 0;
  __builtin_amdgcn_tensor_load_to_lds(g0, g1, z4, z4, z8, 0);
  __builtin_amdgcn_s_wait_tensorcnt((unsigned short)0);
#else
  (void)lds_off; (void)gptr; (void)tile_n; (void)tensorK; (void)tensorN;
#endif
}

// -----------------------------------------------------------------
// Weight prep: W (K x N, f32, row-major) -> Wt (N x K, bf16, row-major)
// -----------------------------------------------------------------
__global__ void prep_weight_kernel(const float* __restrict__ W,
                                   bf16* __restrict__ Wt, int K, int N) {
  int idx = blockIdx.x * blockDim.x + threadIdx.x;
  if (idx >= K * N) return;
  int k = idx / N, n = idx - k * N;
  Wt[(size_t)n * K + k] = (bf16)W[(size_t)k * N + n];
}

// -----------------------------------------------------------------
// Block-tiled WMMA GEMM with TDM-staged B.
//   Block = 256 threads (8 waves). Output tile: 128 rows x NT*16 cols.
//   Wave w owns rows [bm*128 + w*16, +16); all waves share the B strip in LDS.
//   Per 32-K step: wave0 TDM-loads B strip (NT*16 x 32 bf16, 80B pitch),
//   each wave loads its A fragment once and reuses it for NT WMMAs.
// A: M x K (f32 or bf16 row-major); Bt: N x K bf16 (B column-major).
// -----------------------------------------------------------------
template<int NT, bool A_F32, bool OUT_BF16>
__global__ void wmma_gemm_kernel(const void* __restrict__ Av,
                                 const bf16* __restrict__ Bt,
                                 const float* __restrict__ bias,
                                 bf16* __restrict__ Cb,
                                 float* __restrict__ Cf,
                                 int M, int N, int K) {
  __shared__ __align__(16) char BsRaw[NT * 16 * BROW_PITCH];

  const int lane = threadIdx.x & 31;
  const int wave = threadIdx.x >> 5;
  const int nblk = N / (NT * 16);
  const int bm = blockIdx.x / nblk;
  const int bn = blockIdx.x - bm * nblk;
  const int mrow0 = bm * 128 + wave * 16;
  const int n0 = bn * (NT * 16);

  const int kg  = lane >> 4;     // which K-half this lane holds (ISA 7.12.2)
  const int lid = lane & 15;
  const int arow = mrow0 + lid;

  v8f c[NT];
#pragma unroll
  for (int t = 0; t < NT; ++t) c[t] = 0.0f;

  const unsigned lds_off = (unsigned)(size_t)&BsRaw[0];

  for (int kk = 0; kk < K; kk += 32) {
    __syncthreads();                       // previous strip fully consumed
    if (threadIdx.x < 32) {                // wave 0 drives the Tensor Data Mover
      tdm_load_b_tile(lds_off, Bt + (size_t)n0 * K + kk,
                      (unsigned)(NT * 16), (unsigned)K, (unsigned)N);
    }
    __syncthreads();                       // strip visible to all waves

    FragBF a;
    if (A_F32) {
      const float* Ar = (const float*)Av + (size_t)arow * K;
      a.h[0] = cvt8_f32_bf16(Ar + kk + 8 * kg);
      a.h[1] = cvt8_f32_bf16(Ar + kk + 16 + 8 * kg);
    } else {
      const bf16* Ar = (const bf16*)Av + (size_t)arow * K;
      a.h[0] = *(const v8bf*)(Ar + kk + 8 * kg);
      a.h[1] = *(const v8bf*)(Ar + kk + 16 + 8 * kg);
      if (kk + 32 < K) __builtin_prefetch(Ar + kk + 32, 0, 0);  // global_prefetch
    }

#pragma unroll
    for (int t = 0; t < NT; ++t) {
      const bf16* brow = (const bf16*)(BsRaw + (t * 16 + lid) * BROW_PITCH);
      FragBF b;
      b.h[0] = *(const v8bf*)(brow + 8 * kg);        // ds_load_b128
      b.h[1] = *(const v8bf*)(brow + 16 + 8 * kg);
      c[t] = __builtin_amdgcn_wmma_f32_16x16x32_bf16(false, a.v, false, b.v,
                                                     (short)0, c[t], false, false);
    }
  }

  // C/D layout: vgpr r -> row = mrow0 + 8*kg + r, col = n0 + t*16 + lid
  const int orow0 = mrow0 + 8 * kg;
#pragma unroll
  for (int t = 0; t < NT; ++t) {
    const int ocol = n0 + t * 16 + lid;
    const float badd = bias ? bias[ocol] : 0.f;
#pragma unroll
    for (int r = 0; r < 8; ++r) {
      float v = c[t][r] + badd;
      size_t idx = (size_t)(orow0 + r) * N + ocol;
      if (OUT_BF16) Cb[idx] = (bf16)v;
      else          Cf[idx] = v;
    }
  }
}

// -----------------------------------------------------------------
// Depthwise causal 4-tap conv along L + bias + SiLU (xm -> xc, bf16).
// -----------------------------------------------------------------
__global__ void conv_silu_kernel(const bf16* __restrict__ xz,
                                 const float* __restrict__ cw,
                                 const float* __restrict__ cb,
                                 bf16* __restrict__ xc) {
  size_t idx = (size_t)blockIdx.x * blockDim.x + threadIdx.x;
  int d = (int)(idx & (DI - 1));
  int l = (int)((idx >> 9) & (L_ - 1));
  int b = (int)(idx >> 19);
  float acc = cb[d];
#pragma unroll
  for (int j = 0; j < 4; ++j) {
    int li = l - 3 + j;
    if (li >= 0)
      acc = fmaf(cw[d * 4 + j],
                 (float)xz[((size_t)(b * L_ + li)) * (2 * DI) + d], acc);
  }
  float s = acc / (1.f + __expf(-acc));
  xc[idx] = (bf16)s;
}

// -----------------------------------------------------------------
// Fused selective scan: dt projection + softplus + recurrence + D skip
// + z gating + mean-over-L. One block per batch, thread = channel d.
// -----------------------------------------------------------------
#define SCAN_CHUNK 64

__global__ void scan_kernel(const bf16* __restrict__ xc,
                            const bf16* __restrict__ xz,
                            const float* __restrict__ xdbl,   // B*L x 48
                            const float* __restrict__ W_dt,   // 16 x DI
                            const float* __restrict__ b_dt,
                            const float* __restrict__ A_log,  // DI x 16
                            const float* __restrict__ Dp,
                            float* __restrict__ ysum) {
  const int b = blockIdx.x;
  const int d = threadIdx.x;

  float Wdt[RD];
#pragma unroll
  for (int r = 0; r < RD; ++r) Wdt[r] = W_dt[r * DI + d];
  const float bdt = b_dt[d];
  float Avn[NS];
#pragma unroll
  for (int n = 0; n < NS; ++n) Avn[n] = -__expf(A_log[d * NS + n]);
  const float Dd = Dp[d];

  float h[NS];
#pragma unroll
  for (int n = 0; n < NS; ++n) h[n] = 0.f;
  float acc = 0.f;

  __shared__ float sx[SCAN_CHUNK][48];

  for (int l0 = 0; l0 < L_; l0 += SCAN_CHUNK) {
    __syncthreads();
    for (int i = d; i < SCAN_CHUNK * 48; i += DI) {
      int ll = i / 48, cc = i - ll * 48;
      sx[ll][cc] = xdbl[((size_t)(b * L_ + l0 + ll)) * 48 + cc];
    }
    __syncthreads();

    for (int lc = 0; lc < SCAN_CHUNK; ++lc) {
      const int l = l0 + lc;
      const float* row = sx[lc];
      float dtp = bdt;
#pragma unroll
      for (int r = 0; r < RD; ++r) dtp = fmaf(row[r], Wdt[r], dtp);
      float dt = (dtp > 20.f) ? dtp : __logf(1.f + __expf(dtp));

      float x   = (float)xc[((size_t)(b * L_ + l)) * DI + d];
      float dtx = dt * x;

      float y = 0.f;
#pragma unroll
      for (int n = 0; n < NS; ++n) {
        float dA = __expf(dt * Avn[n]);
        h[n] = fmaf(dA, h[n], dtx * row[16 + n]);
        y = fmaf(h[n], row[32 + n], y);
      }
      float z  = (float)xz[((size_t)(b * L_ + l)) * (2 * DI) + DI + d];
      float yf = (y + x * Dd) * (z / (1.f + __expf(-z)));
      acc += yf;
    }
  }
  ysum[b * DI + d] = acc * (1.f / (float)L_);
}

// -----------------------------------------------------------------
// Small dense GEMM head.
// -----------------------------------------------------------------
__global__ void small_gemm_kernel(const float* __restrict__ A,
                                  const float* __restrict__ Bm,
                                  const float* __restrict__ bias,
                                  float* __restrict__ C, int M, int N, int K) {
  int idx = blockIdx.x * blockDim.x + threadIdx.x;
  if (idx >= M * N) return;
  int m = idx / N, n = idx - m * N;
  float acc = bias ? bias[n] : 0.f;
  for (int k = 0; k < K; ++k)
    acc = fmaf(A[(size_t)m * K + k], Bm[(size_t)k * N + n], acc);
  C[idx] = acc;
}

// -----------------------------------------------------------------
extern "C" void kernel_launch(void* const* d_in, const int* in_sizes, int n_in,
                              void* d_out, int out_size, void* d_ws, size_t ws_size,
                              hipStream_t stream) {
  const float* x      = (const float*)d_in[0];
  const float* W_proj = (const float*)d_in[1];
  const float* b_proj = (const float*)d_in[2];
  const float* W_in   = (const float*)d_in[3];
  const float* conv_w = (const float*)d_in[4];
  const float* conv_b = (const float*)d_in[5];
  const float* W_x    = (const float*)d_in[6];
  const float* W_dt   = (const float*)d_in[7];
  const float* b_dt   = (const float*)d_in[8];
  const float* A_log  = (const float*)d_in[9];
  const float* Dparam = (const float*)d_in[10];
  const float* W_out  = (const float*)d_in[11];
  const float* W_c1   = (const float*)d_in[12];
  const float* b_c1   = (const float*)d_in[13];
  const float* W_c2   = (const float*)d_in[14];
  const float* b_c2   = (const float*)d_in[15];
  float* out = (float*)d_out;

  char* base = (char*)d_ws;
  size_t off = 0;
  auto carve = [&](size_t bytes) -> char* {
    char* p = base + off;
    off += (bytes + 255) & ~(size_t)255;
    return p;
  };
  bf16*  wp_t  = (bf16*)carve((size_t)H_ * IND * 2);
  bf16*  wi_t  = (bf16*)carve((size_t)(2 * DI) * H_ * 2);
  bf16*  wx_t  = (bf16*)carve((size_t)48 * DI * 2);
  bf16*  hbuf  = (bf16*)carve((size_t)BL * H_ * 2);
  bf16*  xzbuf = (bf16*)carve((size_t)BL * 2 * DI * 2);
  bf16*  xcbuf = (bf16*)carve((size_t)BL * DI * 2);
  float* xdbl  = (float*)carve((size_t)BL * 48 * 4);
  float* ysum  = (float*)carve((size_t)B_ * DI * 4);
  float* t0    = (float*)carve((size_t)B_ * H_ * 4);
  float* t1    = (float*)carve((size_t)B_ * 128 * 4);

  // 1) weight prep (f32 -> bf16 transposed)
  prep_weight_kernel<<<(IND * H_ + 255) / 256, 256, 0, stream>>>(W_proj, wp_t, IND, H_);
  prep_weight_kernel<<<(H_ * 2 * DI + 255) / 256, 256, 0, stream>>>(W_in, wi_t, H_, 2 * DI);
  prep_weight_kernel<<<(DI * 48 + 255) / 256, 256, 0, stream>>>(W_x, wx_t, DI, 48);

  // 2) h = x @ W_proj_in + b   (M=131072,N=256,K=64), NT=8 -> 128-col strips
  {
    int blocks = (BL / 128) * (H_ / 128);   // 2048
    wmma_gemm_kernel<8, true, true><<<blocks, 256, 0, stream>>>(
        (const void*)x, wp_t, b_proj, hbuf, nullptr, BL, H_, IND);
  }
  // 3) xz = h @ W_in   (M=131072,N=1024,K=256)
  {
    int blocks = (BL / 128) * ((2 * DI) / 128);  // 8192
    wmma_gemm_kernel<8, false, true><<<blocks, 256, 0, stream>>>(
        (const void*)hbuf, wi_t, nullptr, xzbuf, nullptr, BL, 2 * DI, H_);
  }
  // 4) depthwise conv + SiLU
  conv_silu_kernel<<<(int)(((size_t)BL * DI) / 256), 256, 0, stream>>>(
      xzbuf, conv_w, conv_b, xcbuf);

  // 5) x_dbl = xc @ W_x   (M=131072,N=48,K=512), NT=3
  {
    int blocks = (BL / 128) * 1;   // 1024
    wmma_gemm_kernel<3, false, false><<<blocks, 256, 0, stream>>>(
        (const void*)xcbuf, wx_t, nullptr, nullptr, xdbl, BL, 48, DI);
  }
  // 6) fused selective scan + gating + mean pool
  scan_kernel<<<B_, DI, 0, stream>>>(xcbuf, xzbuf, xdbl, W_dt, b_dt, A_log,
                                     Dparam, ysum);
  // 7) head
  small_gemm_kernel<<<(B_ * H_ + 255) / 256, 256, 0, stream>>>(
      ysum, W_out, nullptr, t0, B_, H_, DI);
  small_gemm_kernel<<<(B_ * 128 + 255) / 256, 256, 0, stream>>>(
      t0, W_c1, b_c1, t1, B_, 128, H_);
  small_gemm_kernel<<<(B_ * 12 + 255) / 256, 256, 0, stream>>>(
      t1, W_c2, b_c2, out, B_, 12, 128);
  (void)in_sizes; (void)n_in; (void)out_size; (void)ws_size;
}
